// GCNConvEncoder_55379308315091
// MI455X (gfx1250) — compile-verified, matrix operand order checked by reading, not run on hardware
//
#include <hip/hip_runtime.h>
#include <hip/hip_bf16.h>

// ---------------------------------------------------------------------------
// Types for WMMA bf16 fragments (wave32, v_wmma_f32_16x16x32_bf16)
// ---------------------------------------------------------------------------
typedef __attribute__((ext_vector_type(16))) __bf16 v16bf;
typedef __attribute__((ext_vector_type(8)))  float  v8f;

union Frag32B {
    struct { uint4 lo, hi; } u;   // 32 bytes = 16 bf16
    v16bf v;
};

__device__ __forceinline__ unsigned short f32_to_bf16_rne(float f) {
    unsigned int u = __float_as_uint(f);
    unsigned int r = u + 0x7FFFu + ((u >> 16) & 1u);   // round-to-nearest-even
    return (unsigned short)(r >> 16);
}

// ---------------------------------------------------------------------------
// Normalization kernels
// ---------------------------------------------------------------------------
__global__ void init_deg_kernel(float* deg, int n) {
    int i = blockIdx.x * blockDim.x + threadIdx.x;
    if (i < n) deg[i] = 1.0f;                      // self-loop weight
}

__global__ void edge_deg_kernel(const long long* __restrict__ eidx,
                                const float* __restrict__ w,
                                float* deg, long long E) {
    long long e = (long long)blockIdx.x * blockDim.x + threadIdx.x;
    if (e < E) {
        long long c = eidx[E + e];                 // col
        unsafeAtomicAdd(&deg[c], w[e]);
    }
}

__global__ void dinv_kernel(const float* __restrict__ deg, float* dinv, int n) {
    int i = blockIdx.x * blockDim.x + threadIdx.x;
    if (i < n) {
        float d = deg[i];
        dinv[i] = (d > 0.0f) ? rsqrtf(d) : 0.0f;
    }
}

__global__ void edge_norm_kernel(const long long* __restrict__ eidx,
                                 const float* __restrict__ w,
                                 const float* __restrict__ dinv,
                                 float* __restrict__ norm, long long E) {
    long long e = (long long)blockIdx.x * blockDim.x + threadIdx.x;
    if (e < E) {
        long long r = eidx[e];
        long long c = eidx[E + e];
        norm[e] = dinv[r] * w[e] * dinv[c];
    }
}

__global__ void self_norm_kernel(const float* __restrict__ dinv,
                                 float* __restrict__ ns, int n) {
    int i = blockIdx.x * blockDim.x + threadIdx.x;
    if (i < n) { float d = dinv[i]; ns[i] = d * d; }   // w=1 for self loop
}

// ---------------------------------------------------------------------------
// Conversion kernels
// ---------------------------------------------------------------------------
__global__ void f32_to_bf16_kernel(const float* __restrict__ in,
                                   unsigned short* __restrict__ out, long long n) {
    long long i = (long long)blockIdx.x * blockDim.x + threadIdx.x;
    if (i < n) out[i] = f32_to_bf16_rne(in[i]);
}

// W [K x Nout] row-major fp32  ->  Wt [Nout x K] row-major bf16
__global__ void w_transpose_bf16_kernel(const float* __restrict__ W,
                                        unsigned short* __restrict__ Wt,
                                        int K, int Nout) {
    int i = blockIdx.x * blockDim.x + threadIdx.x;
    if (i < K * Nout) {
        int k = i / Nout, n = i % Nout;
        Wt[(size_t)n * K + k] = f32_to_bf16_rne(W[i]);
    }
}

// ReLU + convert to bf16 (feeds second GEMM)
__global__ void relu_bf16_kernel(const float* __restrict__ in,
                                 unsigned short* __restrict__ out, long long n) {
    long long i = (long long)blockIdx.x * blockDim.x + threadIdx.x;
    if (i < n) out[i] = f32_to_bf16_rne(fmaxf(in[i], 0.0f));
}

// ---------------------------------------------------------------------------
// WMMA bf16 GEMM:  C[M x Nout] = A[M x K] * Bt[Nout x K]^T   (fp32 accumulate)
// One wave per 16x16 output tile; K stepped by 32.
// Fragment layout per CDNA5 ISA 7.12.2 (16-bit A 16x32, f32 C/D 16x16).
// ---------------------------------------------------------------------------
__global__ void gemm_bf16_wmma_kernel(const unsigned short* __restrict__ A,
                                      const unsigned short* __restrict__ Bt,
                                      float* __restrict__ C,
                                      int M, int K, int Nout) {
    const int lane = threadIdx.x & 31;
    const int wave = threadIdx.x >> 5;
    const int tiles_n = Nout >> 4;
    const int total   = (M >> 4) * tiles_n;
    int t = blockIdx.x * 8 + wave;
    if (t >= total) return;
    const int tm = t / tiles_n;
    const int tn = t - tm * tiles_n;

    const int mk   = (tm << 4) + (lane & 15);      // A row for this lane
    const int nk   = (tn << 4) + (lane & 15);      // Bt row (= B column) for this lane
    const int kbase = (lane >> 4) << 3;            // 0 or 8

    const unsigned short* arow = A  + (size_t)mk * K + kbase;
    const unsigned short* brow = Bt + (size_t)nk * K + kbase;

    v8f acc = {};
    for (int k0 = 0; k0 < K; k0 += 32) {
        Frag32B a, b;
        a.u.lo = *(const uint4*)(arow + k0);        // k .. k+7
        a.u.hi = *(const uint4*)(arow + k0 + 16);   // k+16 .. k+23
        b.u.lo = *(const uint4*)(brow + k0);
        b.u.hi = *(const uint4*)(brow + k0 + 16);
        acc = __builtin_amdgcn_wmma_f32_16x16x32_bf16(
                  false, a.v, false, b.v, (short)0, acc, false, false);
    }

    // C/D layout: element r of acc -> row = 8*(lane>=16)+r, col = lane&15
    const int mbase = (tm << 4) + ((lane >> 4) << 3);
    const int ncol  = (tn << 4) + (lane & 15);
    float* crow = C + (size_t)mbase * Nout + ncol;
#pragma unroll
    for (int r = 0; r < 8; ++r) crow[(size_t)r * Nout] = acc[r];
}

// ---------------------------------------------------------------------------
// Aggregation: out = bias + self_loop + sum_e norm[e] * h[row[e]]  (scatter)
// ---------------------------------------------------------------------------
__global__ void agg_init_kernel(const float* __restrict__ h,
                                const float* __restrict__ bias,
                                const float* __restrict__ ns,
                                float* __restrict__ out,
                                int F, long long total) {
    long long i = (long long)blockIdx.x * blockDim.x + threadIdx.x;
    if (i < total) {
        long long node = i / F;
        int f = (int)(i - node * F);
        out[i] = bias[f] + h[i] * ns[node];
    }
}

// One wave32 per edge; each lane handles F/32 floats via float4 + f32 atomics.
template <int F>
__global__ void edge_agg_kernel(const long long* __restrict__ eidx,
                                const float* __restrict__ norm,
                                const float* __restrict__ h,
                                float* __restrict__ out, long long E) {
    const int lane = threadIdx.x & 31;
    long long e = (long long)blockIdx.x * (blockDim.x >> 5) + (threadIdx.x >> 5);
    if (e >= E) return;
    const long long r = eidx[e];
    const long long c = eidx[E + e];
    const float w = norm[e];

    const float4* src = (const float4*)(h   + r * F) + lane;
    float*        dst = (float*)       (out + c * F) + lane * 4;

    constexpr int PER = F / 128;                   // float4s per lane (256->2, 128->1)
#pragma unroll
    for (int j = 0; j < PER; ++j) {
        float4 v = src[j * 32];
        float* d = dst + (size_t)j * 128;
        unsafeAtomicAdd(d + 0, v.x * w);
        unsafeAtomicAdd(d + 1, v.y * w);
        unsafeAtomicAdd(d + 2, v.z * w);
        unsafeAtomicAdd(d + 3, v.w * w);
    }
}

// ---------------------------------------------------------------------------
// Launch
// ---------------------------------------------------------------------------
static inline size_t alignup256(size_t x) { return (x + 255) & ~(size_t)255; }

extern "C" void kernel_launch(void* const* d_in, const int* in_sizes, int n_in,
                              void* d_out, int out_size, void* d_ws, size_t ws_size,
                              hipStream_t stream) {
    constexpr int FIN = 256;
    const float*      x   = (const float*)d_in[0];
    const long long*  ei  = (const long long*)d_in[1];
    const float*      ew  = (const float*)d_in[2];
    const float*      W1  = (const float*)d_in[3];
    const float*      b1  = (const float*)d_in[4];
    const float*      W2  = (const float*)d_in[5];
    const float*      b2  = (const float*)d_in[6];
    float*            out = (float*)d_out;

    const int       N    = in_sizes[0] / FIN;            // 50000
    const long long E    = (long long)in_sizes[2];       // 1.6M
    const int       H    = in_sizes[3] / FIN;            // 256
    const int       FOUT = in_sizes[5] / H;              // 128

    // ---- workspace layout ----
    char* p = (char*)d_ws;
    size_t off = 0;
    auto carve = [&](size_t bytes) { char* q = p + off; off = alignup256(off + bytes); return q; };
    float*          deg    = (float*)         carve((size_t)N * 4);
    float*          dinv   = (float*)         carve((size_t)N * 4);
    float*          norm   = (float*)         carve((size_t)E * 4);
    float*          ns     = (float*)         carve((size_t)N * 4);
    unsigned short* W1t    = (unsigned short*)carve((size_t)FIN * H * 2);
    unsigned short* W2t    = (unsigned short*)carve((size_t)H * FOUT * 2);
    unsigned short* xb     = (unsigned short*)carve((size_t)N * FIN * 2);  // reused as h_bf16
    float*          hg     = (float*)         carve((size_t)N * H * 4);    // GEMM out (reused L2)
    float*          ha     = (float*)         carve((size_t)N * H * 4);    // aggregated layer1
    (void)ws_size;

    const int T = 256;
    auto blks = [&](long long n) { return (unsigned)((n + T - 1) / T); };

    // 1) normalization (cached, shared by both layers)
    init_deg_kernel<<<blks(N), T, 0, stream>>>(deg, N);
    edge_deg_kernel<<<blks(E), T, 0, stream>>>(ei, ew, deg, E);
    dinv_kernel<<<blks(N), T, 0, stream>>>(deg, dinv, N);
    edge_norm_kernel<<<blks(E), T, 0, stream>>>(ei, ew, dinv, norm, E);
    self_norm_kernel<<<blks(N), T, 0, stream>>>(dinv, ns, N);

    // 2) bf16 operands
    f32_to_bf16_kernel<<<blks((long long)N * FIN), T, 0, stream>>>(x, xb, (long long)N * FIN);
    w_transpose_bf16_kernel<<<blks(FIN * H), T, 0, stream>>>(W1, W1t, FIN, H);
    w_transpose_bf16_kernel<<<blks(H * FOUT), T, 0, stream>>>(W2, W2t, H, FOUT);

    // 3) layer 1: GEMM (WMMA) -> init(bias+self) -> edge scatter -> ReLU/bf16
    {
        int tiles = (N >> 4) * (H >> 4);
        gemm_bf16_wmma_kernel<<<(tiles + 7) / 8, T, 0, stream>>>(xb, W1t, hg, N, FIN, H);
    }
    agg_init_kernel<<<blks((long long)N * H), T, 0, stream>>>(hg, b1, ns, ha, H, (long long)N * H);
    edge_agg_kernel<256><<<(unsigned)((E + 7) / 8), T, 0, stream>>>(ei, norm, hg, ha, E);
    relu_bf16_kernel<<<blks((long long)N * H), T, 0, stream>>>(ha, xb, (long long)N * H);

    // 4) layer 2: GEMM (WMMA) -> init into d_out -> edge scatter
    {
        int tiles = (N >> 4) * (FOUT >> 4);
        gemm_bf16_wmma_kernel<<<(tiles + 7) / 8, T, 0, stream>>>(xb, W2t, hg, N, H, FOUT);
    }
    agg_init_kernel<<<blks((long long)N * FOUT), T, 0, stream>>>(hg, b2, ns, out, FOUT, (long long)N * FOUT);
    edge_agg_kernel<128><<<(unsigned)((E + 7) / 8), T, 0, stream>>>(ei, norm, hg, out, E);
}
